// EvoformerBlockV45_80307298500695
// MI455X (gfx1250) — matrix operand order repaired and checked by reading, not compiled, not run on hardware
//
#include <hip/hip_runtime.h>
#include <math.h>

// ---------------------------------------------------------------------------
// CDNA5 (gfx1250) Evoformer block.  All GEMM-shaped math goes through
// V_WMMA_F32_16X16X32_BF16 (f32 data -> bf16 RNE once per LDS tile, f32 acc).
// GEMM requires M%64==0, N%64==0, K%32==0 (true for every call here).
// ---------------------------------------------------------------------------

typedef __attribute__((ext_vector_type(16))) __bf16          v16bf;
typedef __attribute__((ext_vector_type(16))) unsigned short  v16u;
typedef __attribute__((ext_vector_type(8)))  unsigned short  v8u;
typedef __attribute__((ext_vector_type(4)))  unsigned short  v4u;
typedef __attribute__((ext_vector_type(8)))  float           v8f;
typedef __attribute__((ext_vector_type(4)))  float           v4f;

#define DEV __device__ __forceinline__

DEV unsigned short f2bf(float f) {
  unsigned int u = __builtin_bit_cast(unsigned int, f);
  u += 0x7FFFu + ((u >> 16) & 1u);          // round-to-nearest-even
  return (unsigned short)(u >> 16);
}

DEV v4u pk4(v4f f) {
  v4u r;
  r.x = f2bf(f.x); r.y = f2bf(f.y); r.z = f2bf(f.z); r.w = f2bf(f.w);
  return r;
}

DEV v8f wmma_bf16(v16bf a, v16bf b, v8f c) {
  return __builtin_amdgcn_wmma_f32_16x16x32_bf16(false, a, false, b,
                                                 (short)0, c, false, false);
}

union U16 { v16u v; v8u h[2]; };

// ---------------------------------------------------------------------------
// LDS-staged WMMA GEMM (contiguous rows):
//   D[m,n] = act( sum_k A[m,k]*B[k,n] + bias[n] ) * gate[..] + res[m,n]
// 128 threads (4 waves); block tile 64x64; wave tile 16x64; K step 32.
// ---------------------------------------------------------------------------
struct GemmArgs {
  const float* A; long a_rs, a_bs;
  const float* B; long b_rs, b_bs;
  float*       D; long d_rs, d_bs;
  const float* bias;                       // per-n or null
  const float* res;                        // residual, D indexing, or null
  const float* gate; long g_rs, g_cs, g_bs;// multiplier with own strides, or null
  int M, N, K, act;                        // act: 0 none, 1 relu, 2 sigmoid
};

__global__ __launch_bounds__(128) void gemm_wmma_kernel(GemmArgs g) {
  __shared__ unsigned short As[64 * 32];   // [row][k]   bf16 bits
  __shared__ unsigned short Bt[64 * 32];   // [n][k]     bf16 bits (transposed)

  const int tid  = threadIdx.x;
  const int lane = tid & 31;
  const int wave = tid >> 5;               // 0..3
  const int half = lane >> 4;              // 0/1
  const int r    = lane & 15;

  const long z = blockIdx.z;
  const float* A = g.A + z * g.a_bs;
  const float* B = g.B + z * g.b_bs;
  float*       D = g.D + z * g.d_bs;

  const int rm0 = blockIdx.x * 64;         // block row base
  const int cn  = blockIdx.y * 64;         // block col base
  const int ar  = tid >> 1, ac0 = (tid & 1) * 16;  // A staging: 16 elems/thread
  const int kk  = tid >> 2, nn0 = (tid & 3) * 16;  // B staging: 16 elems/thread

  v8f acc[4];
#pragma unroll
  for (int t = 0; t < 4; ++t)
#pragma unroll
    for (int e = 0; e < 8; ++e) acc[t][e] = 0.f;

  for (int kb = 0; kb < g.K; kb += 32) {
    __syncthreads();
    {                                       // stage A 64x32 (b128 loads)
      const float* ap = A + (long)(rm0 + ar) * g.a_rs + (kb + ac0);
      v4f f0 = *(const v4f*)(ap);
      v4f f1 = *(const v4f*)(ap + 4);
      v4f f2 = *(const v4f*)(ap + 8);
      v4f f3 = *(const v4f*)(ap + 12);
      U16 p;
      p.h[0].s0 = f2bf(f0.x); p.h[0].s1 = f2bf(f0.y); p.h[0].s2 = f2bf(f0.z); p.h[0].s3 = f2bf(f0.w);
      p.h[0].s4 = f2bf(f1.x); p.h[0].s5 = f2bf(f1.y); p.h[0].s6 = f2bf(f1.z); p.h[0].s7 = f2bf(f1.w);
      p.h[1].s0 = f2bf(f2.x); p.h[1].s1 = f2bf(f2.y); p.h[1].s2 = f2bf(f2.z); p.h[1].s3 = f2bf(f2.w);
      p.h[1].s4 = f2bf(f3.x); p.h[1].s5 = f2bf(f3.y); p.h[1].s6 = f2bf(f3.z); p.h[1].s7 = f2bf(f3.w);
      *(v16u*)&As[ar * 32 + ac0] = p.v;
      // stage B 32x64 -> Bt[n][k]
      const float* bp = B + (long)(kb + kk) * g.b_rs + (cn + nn0);
      v4f g0 = *(const v4f*)(bp);
      v4f g1 = *(const v4f*)(bp + 4);
      v4f g2 = *(const v4f*)(bp + 8);
      v4f g3 = *(const v4f*)(bp + 12);
      unsigned short bb[16];
      bb[0]=f2bf(g0.x); bb[1]=f2bf(g0.y); bb[2]=f2bf(g0.z); bb[3]=f2bf(g0.w);
      bb[4]=f2bf(g1.x); bb[5]=f2bf(g1.y); bb[6]=f2bf(g1.z); bb[7]=f2bf(g1.w);
      bb[8]=f2bf(g2.x); bb[9]=f2bf(g2.y); bb[10]=f2bf(g2.z); bb[11]=f2bf(g2.w);
      bb[12]=f2bf(g3.x); bb[13]=f2bf(g3.y); bb[14]=f2bf(g3.z); bb[15]=f2bf(g3.w);
#pragma unroll
      for (int i = 0; i < 16; ++i) Bt[(nn0 + i) * 32 + kk] = bb[i];
    }
    __syncthreads();

    const unsigned short* arow = &As[(wave * 16 + r) * 32];
    U16 ua;
    ua.h[0] = *(const v8u*)(arow + half * 8);         // K 0-7 / 8-15
    ua.h[1] = *(const v8u*)(arow + 16 + half * 8);    // K 16-23 / 24-31
    v16bf af = __builtin_bit_cast(v16bf, ua.v);
#pragma unroll
    for (int t = 0; t < 4; ++t) {
      U16 bu;
      bu.h[0] = *(const v8u*)&Bt[(t * 16 + r) * 32 + half * 8];
      bu.h[1] = *(const v8u*)&Bt[(t * 16 + r) * 32 + 16 + half * 8];
      acc[t] = wmma_bf16(af, __builtin_bit_cast(v16bf, bu.v), acc[t]);
    }
  }

  const float* res  = g.res  ? g.res  + z * g.d_bs : nullptr;
  const float* gate = g.gate ? g.gate + z * g.g_bs : nullptr;
#pragma unroll
  for (int t = 0; t < 4; ++t) {
    int n = cn + t * 16 + r;
    float bs = g.bias ? g.bias[n] : 0.f;
#pragma unroll
    for (int v = 0; v < 8; ++v) {
      int m = rm0 + wave * 16 + half * 8 + v;         // C/D: VGPR -> M
      long idx = (long)m * g.d_rs + n;
      float x = acc[t][v] + bs;
      if (g.act == 1)      x = x > 0.f ? x : 0.f;
      else if (g.act == 2) x = 1.f / (1.f + __expf(-x));
      if (gate) x *= gate[(long)m * g.g_rs + (long)n * g.g_cs];
      if (res)  x += res[idx];
      D[idx] = x;
    }
  }
}

// ---------------------------------------------------------------------------
// Fused attention: one workgroup (256 thr / 8 waves) per (group g, head h).
// qkv rows 768 wide: [q|k|v], channel = sel*256 + h*32 + d, dh = 32.
// LDS: f32 scores LxL + bf16-staged Q,K (row major) and V (transposed).
// L=256: 256KB + 48KB = 304KB <= 320KB WGP LDS.
// ---------------------------------------------------------------------------
__global__ __launch_bounds__(256) void attn_kernel(
    const float* __restrict__ qkv, long grpStride,
    const float* __restrict__ bias, long b_is, long b_js, long b_hs,
    float* __restrict__ out, long out_grp,
    int L, int H, float scale) {
  extern __shared__ char smem[];
  float* sc           = (float*)smem;                       // L*L scores
  unsigned short* qb  = (unsigned short*)(smem + (size_t)L * L * 4); // [i][c] L*32
  unsigned short* kbf = qb + L * 32;                        // [j][c]
  unsigned short* vt  = kbf + L * 32;                       // [d][m] 32*L

  const int g    = blockIdx.x / H;
  const int h    = blockIdx.x % H;
  const int lane = threadIdx.x & 31;
  const int wave = threadIdx.x >> 5;       // 0..7
  const int half = lane >> 4;
  const int r    = lane & 15;

  const float* Q = qkv + (long)g * grpStride + h * 32;
  const float* K = Q + 256;
  const float* V = Q + 512;
  const int T = L / 16;

  // phase 0: stage Q,K,V into LDS as bf16
  for (int e = threadIdx.x * 4; e < L * 32; e += blockDim.x * 4) {
    int row = e >> 5, c = e & 31;
    v4f q4 = *(const v4f*)(Q + (long)row * 768 + c);
    v4f k4 = *(const v4f*)(K + (long)row * 768 + c);
    v4f w4 = *(const v4f*)(V + (long)row * 768 + c);
    *(v4u*)&qb[e]  = pk4(q4);
    *(v4u*)&kbf[e] = pk4(k4);
    vt[(c + 0) * L + row] = f2bf(w4.x);
    vt[(c + 1) * L + row] = f2bf(w4.y);
    vt[(c + 2) * L + row] = f2bf(w4.z);
    vt[(c + 3) * L + row] = f2bf(w4.w);
  }
  __syncthreads();

  // phase 1: S = scale * Q K^T + bias
  for (int t = wave; t < T * T; t += 8) {
    int ti = t / T, tj = t % T;
    const unsigned short* qrow = &qb[(ti * 16 + r) * 32];
    U16 ua;
    ua.h[0] = *(const v8u*)(qrow + half * 8);
    ua.h[1] = *(const v8u*)(qrow + 16 + half * 8);
    U16 bu;
    bu.h[0] = *(const v8u*)&kbf[(tj * 16 + r) * 32 + half * 8];
    bu.h[1] = *(const v8u*)&kbf[(tj * 16 + r) * 32 + 16 + half * 8];
    v8f acc;
#pragma unroll
    for (int e = 0; e < 8; ++e) acc[e] = 0.f;
    acc = wmma_bf16(__builtin_bit_cast(v16bf, ua.v),
                    __builtin_bit_cast(v16bf, bu.v), acc);
#pragma unroll
    for (int v = 0; v < 8; ++v) {
      int i = ti * 16 + half * 8 + v;
      int j = tj * 16 + r;
      float x = acc[v] * scale;
      if (bias) x += bias[(long)i * b_is + (long)j * b_js + (long)h * b_hs];
      sc[(long)i * L + j] = x;
    }
  }
  __syncthreads();

  // phase 2: row softmax in LDS
  for (int i = threadIdx.x; i < L; i += blockDim.x) {
    float* rowp = sc + (long)i * L;
    float mx = -1e30f;
    for (int j = 0; j < L; ++j) mx = fmaxf(mx, rowp[j]);
    float sum = 0.f;
    for (int j = 0; j < L; ++j) { float e = __expf(rowp[j] - mx); rowp[j] = e; sum += e; }
    float inv = 1.f / sum;
    for (int j = 0; j < L; ++j) rowp[j] *= inv;
  }
  __syncthreads();

  // phase 3: O = P V  (two 16-wide column tiles)
  for (int t = wave; t < T * 2; t += 8) {
    int ti = t >> 1, tc = t & 1;
    v8f acc;
#pragma unroll
    for (int e = 0; e < 8; ++e) acc[e] = 0.f;
    for (int kb = 0; kb < L; kb += 32) {
      const float* prow = sc + (long)(ti * 16 + r) * L + kb;
      v4f p0 = *(const v4f*)(prow + half * 8);
      v4f p1 = *(const v4f*)(prow + half * 8 + 4);
      v4f p2 = *(const v4f*)(prow + 16 + half * 8);
      v4f p3 = *(const v4f*)(prow + 16 + half * 8 + 4);
      U16 ua;
      ua.h[0].s0=f2bf(p0.x); ua.h[0].s1=f2bf(p0.y); ua.h[0].s2=f2bf(p0.z); ua.h[0].s3=f2bf(p0.w);
      ua.h[0].s4=f2bf(p1.x); ua.h[0].s5=f2bf(p1.y); ua.h[0].s6=f2bf(p1.z); ua.h[0].s7=f2bf(p1.w);
      ua.h[1].s0=f2bf(p2.x); ua.h[1].s1=f2bf(p2.y); ua.h[1].s2=f2bf(p2.z); ua.h[1].s3=f2bf(p2.w);
      ua.h[1].s4=f2bf(p3.x); ua.h[1].s5=f2bf(p3.y); ua.h[1].s6=f2bf(p3.z); ua.h[1].s7=f2bf(p3.w);
      U16 bu;
      bu.h[0] = *(const v8u*)&vt[(tc * 16 + r) * L + kb + half * 8];
      bu.h[1] = *(const v8u*)&vt[(tc * 16 + r) * L + kb + 16 + half * 8];
      acc = wmma_bf16(__builtin_bit_cast(v16bf, ua.v),
                      __builtin_bit_cast(v16bf, bu.v), acc);
    }
#pragma unroll
    for (int v = 0; v < 8; ++v) {
      int i = ti * 16 + half * 8 + v;
      out[(long)g * out_grp + (long)i * 256 + h * 32 + tc * 16 + r] = acc[v];
    }
  }
}

// ---------------------------------------------------------------------------
// LayerNorm over last dim C (blockDim == C, C in {128,256}).
// ---------------------------------------------------------------------------
__global__ void ln_kernel(const float* __restrict__ in, float* __restrict__ out,
                          const float* __restrict__ gam, const float* __restrict__ bet,
                          int C) {
  __shared__ float s1[256], s2[256];
  long row = blockIdx.x;
  int t = threadIdx.x;
  float v = in[row * C + t];
  s1[t] = v; s2[t] = v * v;
  __syncthreads();
  for (int off = C >> 1; off > 0; off >>= 1) {
    if (t < off) { s1[t] += s1[t + off]; s2[t] += s2[t + off]; }
    __syncthreads();
  }
  float mean = s1[0] / C;
  float var  = s2[0] / C - mean * mean;
  float inv  = __frsqrt_rn(var + 1e-5f);
  out[row * C + t] = (v - mean) * inv * gam[t] + bet[t];
}

// out[(d1*D0+d0)*C+c] = in[(d0*D1+d1)*C+c]
__global__ void transpose_kernel(const float* __restrict__ in, float* __restrict__ out,
                                 int D0, int D1, int C) {
  long i = (long)blockIdx.x * blockDim.x + threadIdx.x;
  long total = (long)D0 * D1 * C;
  if (i >= total) return;
  int c = (int)(i % C);
  long ij = i / C;
  int d1 = (int)(ij % D1);
  int d0 = (int)(ij / D1);
  out[((long)d1 * D0 + d0) * C + c] = in[i];
}

__global__ void mean_s_kernel(const float* __restrict__ msa, float* __restrict__ mean,
                              int S, int N, int C) {
  long i = (long)blockIdx.x * blockDim.x + threadIdx.x;
  if (i >= (long)N * C) return;
  int c = (int)(i % C);
  int n = (int)(i / C);
  float acc = 0.f;
  for (int s = 0; s < S; ++s) acc += msa[((long)s * N + n) * C + c];
  mean[i] = acc / (float)S;
}

// pair bias projection: out[ij,h] = pair[ij,:]@w[:,h] + b[h]   (N=8 too small for WMMA)
__global__ void pb_kernel(const float* __restrict__ pair, const float* __restrict__ w,
                          const float* __restrict__ b, float* __restrict__ out) {
  long idx = (long)blockIdx.x * blockDim.x + threadIdx.x;   // 65536*8
  int h = (int)(idx & 7);
  long ij = idx >> 3;
  const float* row = pair + ij * 128;
  float acc = b[h];
  for (int k = 0; k < 128; ++k) acc += row[k] * w[k * 8 + h];
  out[idx] = acc;
}

// pair_out[i,j,:] = pair_in[i,j,:] + LN(lr[i,:]*lr[j,:])   (blockDim == C == 128)
__global__ void outer_ln_add_kernel(const float* __restrict__ lr,
                                    const float* __restrict__ pair_in,
                                    float* __restrict__ out,
                                    const float* __restrict__ gam,
                                    const float* __restrict__ bet,
                                    int N, int C) {
  __shared__ float s1[128], s2[128];
  int bx = blockIdx.x;
  int i = bx / N, j = bx % N;
  int t = threadIdx.x;
  float v = lr[(long)i * C + t] * lr[(long)j * C + t];
  s1[t] = v; s2[t] = v * v;
  __syncthreads();
  for (int off = C >> 1; off > 0; off >>= 1) {
    if (t < off) { s1[t] += s1[t + off]; s2[t] += s2[t + off]; }
    __syncthreads();
  }
  float mean = s1[0] / C;
  float var  = s2[0] / C - mean * mean;
  float inv  = __frsqrt_rn(var + 1e-5f);
  long idx = (long)bx * C + t;
  out[idx] = pair_in[idx] + ((v - mean) * inv * gam[t] + bet[t]);
}

// ---------------------------------------------------------------------------
// Host side
// ---------------------------------------------------------------------------
static void gemm(hipStream_t st,
                 const float* A, long ars, long abs_,
                 const float* B, long brs, long bbs,
                 float* D, long drs, long dbs,
                 const float* bias, const float* res,
                 const float* gate, long grs, long gcs, long gbs,
                 int M, int N, int K, int act, int batch) {
  GemmArgs g{A, ars, abs_, B, brs, bbs, D, drs, dbs,
             bias, res, gate, grs, gcs, gbs, M, N, K, act};
  dim3 grid(M / 64, N / 64, batch);
  gemm_wmma_kernel<<<grid, 128, 0, st>>>(g);
}

extern "C" void kernel_launch(void* const* d_in, const int* in_sizes, int n_in,
                              void* d_out, int out_size, void* d_ws, size_t ws_size,
                              hipStream_t stream) {
  (void)in_sizes; (void)n_in; (void)out_size; (void)ws_size;
  const int S = 64, N = 256, DIM = 256, CP = 128, H = 8;
  const float scale = 1.f / sqrtf(32.f);

  const float* msa_in  = (const float*)d_in[0];
  const float* pair_in = (const float*)d_in[1];
#define PI(i) ((const float*)d_in[i])

  float* out_msa  = (float*)d_out;            // 64*256*256
  float* out_pair = out_msa + 4194304;        // 256*256*128

  float* ws = (float*)d_ws;
  size_t off = 0;
  float* qkvb  = ws + off; off += 12582912;   // 16384*768  (qkv / rightT / tr scratch)
  float* B1    = ws + off; off += 4194304;
  float* B2    = ws + off; off += 4194304;
  float* B3    = ws + off; off += 4194304;
  float* pairb = ws + off; off += 8388608;    // evolving pair state
  float* pA    = ws + off; off += 8388608;    // pA..pD contiguous -> 65536*512 hidden
  float* pB    = ws + off; off += 8388608;
  float* pC    = ws + off; off += 8388608;
  float* pD    = ws + off; off += 8388608;
  float* pbb   = ws + off; off += 524288;     // pair bias 256*256*8
  float* meanb = ws + off; off += 65536;
  float* lrb   = ws + off; off += 32768;

  const long NL = 0;
  // ---------------- row attention ----------------
  gemm(stream, msa_in, DIM, 0, PI(2), 3 * DIM, 0, qkvb, 3 * DIM, 0,
       PI(3), nullptr, nullptr, NL, NL, NL, S * N, 3 * DIM, DIM, 0, 1);   // qkv
  pb_kernel<<<(N * N * H) / 256, 256, 0, stream>>>(pair_in, PI(8), PI(9), pbb);
  attn_kernel<<<S * H, 256, (size_t)N * N * 4 + 3 * N * 64, stream>>>(
      qkvb, (long)N * 768, pbb, (long)N * H, H, 1, B1, (long)N * DIM, N, H, scale);
  gemm(stream, B1, DIM, 0, PI(4), DIM, 0, B2, DIM, 0,
       PI(5), msa_in, nullptr, NL, NL, NL, S * N, DIM, DIM, 0, 1);        // out + res
  ln_kernel<<<S * N, DIM, 0, stream>>>(B2, B3, PI(6), PI(7), DIM);        // msa1

  // ---------------- column attention ----------------
  transpose_kernel<<<(S * N * DIM + 255) / 256, 256, 0, stream>>>(B3, B1, S, N, DIM);
  gemm(stream, B1, DIM, 0, PI(10), 3 * DIM, 0, qkvb, 3 * DIM, 0,
       PI(11), nullptr, nullptr, NL, NL, NL, S * N, 3 * DIM, DIM, 0, 1);  // col qkv
  attn_kernel<<<N * H, 256, (size_t)S * S * 4 + 3 * S * 64, stream>>>(
      qkvb, (long)S * 768, nullptr, 0, 0, 0, B2, (long)S * DIM, S, H, scale);
  transpose_kernel<<<(S * N * DIM + 255) / 256, 256, 0, stream>>>(B2, B1, N, S, DIM);
  gemm(stream, B1, DIM, 0, PI(12), DIM, 0, B2, DIM, 0,
       PI(13), B3, nullptr, NL, NL, NL, S * N, DIM, DIM, 0, 1);
  ln_kernel<<<S * N, DIM, 0, stream>>>(B2, out_msa, PI(14), PI(15), DIM); // final msa

  // ---------------- outer product update ----------------
  mean_s_kernel<<<(N * DIM + 255) / 256, 256, 0, stream>>>(out_msa, meanb, S, N, DIM);
  gemm(stream, meanb, DIM, 0, PI(16), CP, 0, lrb, CP, 0,
       PI(17), nullptr, nullptr, NL, NL, NL, N, CP, DIM, 0, 1);
  outer_ln_add_kernel<<<N * N, CP, 0, stream>>>(lrb, pair_in, pairb, PI(18), PI(19), N, CP);

  // ---------------- triangle blocks ----------------
  for (int tb = 0; tb < 2; ++tb) {
    // left = lin(ln(pair), left_w) [+ lin(pair, eq_w) for t1]  -> pB [ik,h]
    if (tb == 0) {
      gemm(stream, pairb, CP, 0, PI(30), CP, 0, pB, CP, 0,
           PI(31), nullptr, nullptr, NL, NL, NL, N * N, CP, CP, 0, 1);    // eq term
      ln_kernel<<<N * N, CP, 0, stream>>>(pairb, pA, PI(20), PI(21), CP);
      gemm(stream, pA, CP, 0, PI(24), CP, 0, pB, CP, 0,
           PI(25), pB, nullptr, NL, NL, NL, N * N, CP, CP, 0, 1);
    } else {
      ln_kernel<<<N * N, CP, 0, stream>>>(pairb, pA, PI(36), PI(37), CP);
      gemm(stream, pA, CP, 0, PI(40), CP, 0, pB, CP, 0,
           PI(41), nullptr, nullptr, NL, NL, NL, N * N, CP, CP, 0, 1);
    }
    // right -> pC [kj,h]
    ln_kernel<<<N * N, CP, 0, stream>>>(pairb, pA, PI(tb ? 38 : 22), PI(tb ? 39 : 23), CP);
    gemm(stream, pA, CP, 0, PI(tb ? 42 : 26), CP, 0, pC, CP, 0,
         PI(tb ? 43 : 27), nullptr, nullptr, NL, NL, NL, N * N, CP, CP, 0, 1);
    // gate = sigmoid(lin(pair)) -> pA [ij,h]
    gemm(stream, pairb, CP, 0, PI(tb ? 44 : 28), CP, 0, pA, CP, 0,
         PI(tb ? 45 : 29), nullptr, nullptr, NL, NL, NL, N * N, CP, CP, 2, 1);
    // re-layout to head-major for contiguous batched GEMM
    transpose_kernel<<<(N * N * CP + 255) / 256, 256, 0, stream>>>(pB, pD, N * N, CP, 1);   // leftT [h,ik]
    transpose_kernel<<<(N * N * CP + 255) / 256, 256, 0, stream>>>(pC, qkvb, N * N, CP, 1); // rightT [h,kj]
    // mulT[h,i,j] = (sum_k leftT[h,i,k]*rightT[h,k,j]) * gate[i,j,h]
    gemm(stream, pD, N, (long)N * N, qkvb, N, (long)N * N, pB, N, (long)N * N,
         nullptr, nullptr, pA, (long)N * CP, CP, 1, N, N, N, 0, CP);
    transpose_kernel<<<(N * N * CP + 255) / 256, 256, 0, stream>>>(pB, pC, CP, N * N, 1);   // mul [ij,h]
    // out proj + residual, then LN -> pair
    gemm(stream, pC, CP, 0, PI(tb ? 46 : 32), CP, 0, pD, CP, 0,
         PI(tb ? 47 : 33), pairb, nullptr, NL, NL, NL, N * N, CP, CP, 0, 1);
    ln_kernel<<<N * N, CP, 0, stream>>>(pD, pairb, PI(tb ? 48 : 34), PI(tb ? 49 : 35), CP);
  }

  // ---------------- transition ----------------
  gemm(stream, pairb, CP, 0, PI(50), 4 * CP, 0, pA, 4 * CP, 0,
       PI(51), nullptr, nullptr, NL, NL, NL, N * N, 4 * CP, CP, 1, 1);    // relu 65536x512
  gemm(stream, pA, 4 * CP, 0, PI(52), CP, 0, qkvb, CP, 0,
       PI(53), pairb, nullptr, NL, NL, NL, N * N, CP, 4 * CP, 0, 1);
  ln_kernel<<<N * N, CP, 0, stream>>>(qkvb, pairb, PI(54), PI(55), CP);
  ln_kernel<<<N * N, CP, 0, stream>>>(pairb, out_pair, PI(56), PI(57), CP);
#undef PI
}
